// ParallelRetention_59639915872873
// MI455X (gfx1250) — matrix operand order repaired and checked by reading, not compiled, not run on hardware
//
#include <hip/hip_runtime.h>

// ---------------------------------------------------------------------------
// Types for gfx1250 WMMA (wave32)
// ---------------------------------------------------------------------------
typedef __attribute__((ext_vector_type(16))) _Float16 v16h;
typedef __attribute__((ext_vector_type(8)))  _Float16 v8h;
typedef __attribute__((ext_vector_type(8)))  float    v8f;

union V16 { v16h v; v8h h[2]; };

// v_wmma_f32_16x16x32_f16 : D = A(16x32 f16) * B(32x16 f16) + C(16x16 f32)
#define WMMA_F16(a, b, c) \
  __builtin_amdgcn_wmma_f32_16x16x32_f16(false, (a), false, (b), (short)0, (c), false, false)

__device__ __forceinline__ float gelu_exact(float x) {
  return 0.5f * x * (1.0f + erff(x * 0.7071067811865475f));
}

// Async global->LDS copy (16B per lane), tracked by ASYNCcnt.
// ldsOff: 32-bit LDS byte offset (low 32 bits of a __shared__ pointer),
// gptr  : 64-bit global address (VGPR pair).
__device__ __forceinline__ void async_copy_b128(unsigned ldsOff, const void* gptr) {
  asm volatile("global_load_async_to_lds_b128 %0, %1, off"
               :: "v"(ldsOff), "v"(gptr) : "memory");
}
__device__ __forceinline__ void wait_asynccnt0() {
  asm volatile("s_wait_asynccnt 0" ::: "memory");
}

// ---------------------------------------------------------------------------
// Generic WMMA GEMM:  C[M,N] = act(A[M,K] @ W[K,N] + bias)
//   A: float or _Float16 (row-major, leading dim lda)
//   W: float  (row-major, leading dim ldw == N)
//   C: float or _Float16 (row-major, leading dim ldc)
// Block: 256 threads (8 waves), tile BM=128 x BN=128, BK=64.
// Wave grid 4(M) x 2(N): each wave owns a 32x64 tile = 2x4 WMMA tiles.
// ---------------------------------------------------------------------------
template <typename AT, typename OT, bool GELU>
__global__ __launch_bounds__(256) void gemm_wmma_kernel(
    const AT* __restrict__ A, int lda,
    const float* __restrict__ W, int ldw,
    const float* __restrict__ bias,
    OT* __restrict__ C, int ldc,
    int M, int N, int K)
{
  constexpr int BM = 128, BN = 128, BK = 64, LP = BK + 8;  // pad rows to 72 halves
  __shared__ _Float16 As[BM * LP];   // As[m][k]
  __shared__ _Float16 Bs[BN * LP];   // Bs[n][k]  (W transposed)

  const int tid  = threadIdx.x;
  const int lane = tid & 31;
  const int wid  = tid >> 5;
  const int wm   = wid & 3;          // 0..3  (M direction)
  const int wn   = wid >> 2;         // 0..1  (N direction)
  const int rowBase = blockIdx.y * BM;
  const int colBase = blockIdx.x * BN;

  v8f acc[2][4] = {};

  for (int kb = 0; kb < K; kb += BK) {
    __syncthreads();
    // ---- load A tile (convert f32->f16 if needed) ----
    if constexpr (sizeof(AT) == 4) {
      #pragma unroll
      for (int p = 0; p < 8; ++p) {
        const int row = p * 16 + (tid >> 4);
        const int col = (tid & 15) * 4;
        const float4 x = *(const float4*)(A + (size_t)(rowBase + row) * lda + kb + col);
        _Float16* d = As + row * LP + col;
        d[0] = (_Float16)x.x; d[1] = (_Float16)x.y;
        d[2] = (_Float16)x.z; d[3] = (_Float16)x.w;
      }
    } else {
      #pragma unroll
      for (int p = 0; p < 4; ++p) {
        const int row = p * 32 + (tid >> 3);
        const int col = (tid & 7) * 8;
        *(v8h*)(As + row * LP + col) =
            *(const v8h*)((const _Float16*)A + (size_t)(rowBase + row) * lda + kb + col);
      }
    }
    // ---- load W tile transposed: Bs[n][k] = W[kb+k][colBase+n] ----
    #pragma unroll
    for (int p = 0; p < 8; ++p) {
      const int k  = p * 8 + (tid >> 5);
      const int n0 = (tid & 31) * 4;
      const float4 x = *(const float4*)(W + (size_t)(kb + k) * ldw + colBase + n0);
      Bs[(n0 + 0) * LP + k] = (_Float16)x.x;
      Bs[(n0 + 1) * LP + k] = (_Float16)x.y;
      Bs[(n0 + 2) * LP + k] = (_Float16)x.z;
      Bs[(n0 + 3) * LP + k] = (_Float16)x.w;
    }
    __syncthreads();

    #pragma unroll
    for (int kc = 0; kc < BK; kc += 32) {
      // A fragments: lane<16 -> row m=lane, K {kc..kc+7, kc+16..23}; lane>=16 -> K {+8,+24}
      V16 a[2];
      #pragma unroll
      for (int mi = 0; mi < 2; ++mi) {
        const int m = wm * 32 + mi * 16 + (lane & 15);
        const _Float16* p = As + m * LP + kc + (lane >> 4) * 8;
        a[mi].h[0] = *(const v8h*)p;
        a[mi].h[1] = *(const v8h*)(p + 16);
      }
      #pragma unroll
      for (int ni = 0; ni < 4; ++ni) {
        // B fragment: lane -> col n, K {kc+(lane>>4)*8 ..+7, +16..+23}
        const int n = wn * 64 + ni * 16 + (lane & 15);
        const _Float16* p = Bs + n * LP + kc + (lane >> 4) * 8;
        V16 b;
        b.h[0] = *(const v8h*)p;
        b.h[1] = *(const v8h*)(p + 16);
        #pragma unroll
        for (int mi = 0; mi < 2; ++mi)
          acc[mi][ni] = WMMA_F16(a[mi].v, b.v, acc[mi][ni]);
      }
    }
  }

  // ---- epilogue: bias (+gelu) + store ----
  #pragma unroll
  for (int mi = 0; mi < 2; ++mi)
    #pragma unroll
    for (int ni = 0; ni < 4; ++ni) {
      const int gc = colBase + wn * 64 + ni * 16 + (lane & 15);
      const float bv = bias[gc];
      #pragma unroll
      for (int v = 0; v < 8; ++v) {
        const int gr = rowBase + wm * 32 + mi * 16 + (lane >> 4) * 8 + v;
        float x = acc[mi][ni][v] + bv;
        if constexpr (GELU) x = gelu_exact(x);
        C[(size_t)gr * ldc + gc] = (OT)x;
      }
    }
}

// ---------------------------------------------------------------------------
// Fused retention:  h_ret = (D ∘ (Q @ K^T)) @ V   without materializing scores.
// Trick: compute S^T = K @ Q^T ; the WMMA C-layout of S^T equals the A-layout
// of S for the second WMMA, so decay-multiply + f32->f16 happens in registers.
// Block: 256 threads (8 waves). 64 q-rows per WG; waves 4(M) x 2(inter-half).
// Per j-step: 32 kv rows; K tile streamed in two 256-wide chunks via
// GLOBAL_LOAD_ASYNC_TO_LDS_B128 (ASYNCcnt), V^T staged through registers.
// D rows prefetched one j-block ahead (global_prefetch_b8).
// ---------------------------------------------------------------------------
__global__ __launch_bounds__(256) void retention_kernel(
    const _Float16* __restrict__ Q, const _Float16* __restrict__ K,
    const _Float16* __restrict__ V, const float* __restrict__ D,
    _Float16* __restrict__ O, int Nrows)
{
  constexpr int KP = 264;  // 256 + 8 pad (halves)
  constexpr int VP = 40;   // 32 + 8 pad (halves)
  __shared__ _Float16 Ks[32 * KP];    // Ks[kv][f_local]           (16.9 KB)
  __shared__ _Float16 Vst[512 * VP];  // Vst[n][kv]  (V transposed) (41.0 KB)

  const int tid  = threadIdx.x;
  const int lane = tid & 31;
  const int wid  = tid >> 5;
  const int wm   = wid & 3;           // q sub-block (16 rows)
  const int wn   = wid >> 2;          // inter half (256 cols)
  const int rowBase = blockIdx.x * 64;
  const int q0   = wm * 16;
  const int nb   = wn * 256;

  // Per-thread K-chunk copy slot (16B): row r, col c within 32 x 256 chunk.
  const int kr = tid >> 5;            // 0..7 (stepped by 8 over 4 passes)
  const int kc8 = (tid & 31) * 8;     // halves

  // Preload this wave's Q B-fragments (Q^T) from global: 16 frags = 128 VGPRs.
  const int qrow = rowBase + q0 + (lane & 15);
  V16 qb[16];
  #pragma unroll
  for (int f = 0; f < 16; ++f) {
    const _Float16* p = Q + (size_t)qrow * 512 + f * 32 + (lane >> 4) * 8;
    qb[f].h[0] = *(const v8h*)p;
    qb[f].h[1] = *(const v8h*)(p + 16);
  }

  v8f o[16] = {};  // 16 q-rows x 256 inter cols

  for (int jb = 0; jb < Nrows; jb += 32) {
    __syncthreads();
    // ---- K tile chunk 0 (features 0..255): async global->LDS ----
    #pragma unroll
    for (int p = 0; p < 4; ++p) {
      const int r = p * 8 + kr;
      async_copy_b128((unsigned)(size_t)(Ks + r * KP + kc8),
                      K + (size_t)(jb + r) * 512 + kc8);
    }
    // ---- V tile transposed into LDS: Vst[n][kv] = V[jb+kv][n] ----
    {
      const int kv = tid >> 3, nblk = tid & 7;
      #pragma unroll
      for (int p = 0; p < 8; ++p) {
        const int n0 = nblk * 8 + p * 64;
        const v8h vv = *(const v8h*)(V + (size_t)(jb + kv) * 512 + n0);
        #pragma unroll
        for (int j = 0; j < 8; ++j) Vst[(n0 + j) * VP + kv] = vv[j];
      }
    }
    wait_asynccnt0();     // K chunk 0 resident in LDS before barrier release
    __syncthreads();

    v8f t0 = {}, t1 = {};  // S^T tiles: kv rows 0..15 / 16..31, cols = 16 q rows
    #pragma unroll
    for (int fc = 0; fc < 8; ++fc) {
      const int fl = fc * 32;
      V16 a0, a1;
      const _Float16* p0 = Ks + (lane & 15) * KP + fl + (lane >> 4) * 8;
      const _Float16* p1 = p0 + 16 * KP;
      a0.h[0] = *(const v8h*)p0; a0.h[1] = *(const v8h*)(p0 + 16);
      a1.h[0] = *(const v8h*)p1; a1.h[1] = *(const v8h*)(p1 + 16);
      t0 = WMMA_F16(a0.v, qb[fc].v, t0);
      t1 = WMMA_F16(a1.v, qb[fc].v, t1);
    }
    __syncthreads();
    // ---- K tile chunk 1 (features 256..511): async global->LDS ----
    #pragma unroll
    for (int p = 0; p < 4; ++p) {
      const int r = p * 8 + kr;
      async_copy_b128((unsigned)(size_t)(Ks + r * KP + kc8),
                      K + (size_t)(jb + r) * 512 + 256 + kc8);
    }
    wait_asynccnt0();
    __syncthreads();
    #pragma unroll
    for (int fc = 8; fc < 16; ++fc) {
      const int fl = (fc - 8) * 32;
      V16 a0, a1;
      const _Float16* p0 = Ks + (lane & 15) * KP + fl + (lane >> 4) * 8;
      const _Float16* p1 = p0 + 16 * KP;
      a0.h[0] = *(const v8h*)p0; a0.h[1] = *(const v8h*)(p0 + 16);
      a1.h[0] = *(const v8h*)p1; a1.h[1] = *(const v8h*)(p1 + 16);
      t0 = WMMA_F16(a0.v, qb[fc].v, t0);
      t1 = WMMA_F16(a1.v, qb[fc].v, t1);
    }

    // ---- decay multiply (S^T C-layout == S A-layout) + cvt to f16 ----
    const float* Drow = D + (size_t)qrow * Nrows + jb + 8 * (lane >> 4);
    if (jb + 32 < Nrows)
      __builtin_prefetch(Drow + 32, 0, 3);   // next j-block's D lines -> GL2
    V16 s;
    #pragma unroll
    for (int v = 0; v < 8; ++v) {
      s.v[v]     = (_Float16)(t0[v] * Drow[v]);        // kv = 8*half + v
      s.v[8 + v] = (_Float16)(t1[v] * Drow[16 + v]);   // kv = 8*half + v + 16
    }

    // ---- O += S(16x32) @ V(32x256) ----
    #pragma unroll
    for (int nt = 0; nt < 16; ++nt) {
      const int n = nb + nt * 16 + (lane & 15);
      const _Float16* pv = Vst + n * VP + (lane >> 4) * 8;
      V16 b;
      b.h[0] = *(const v8h*)pv;
      b.h[1] = *(const v8h*)(pv + 16);
      o[nt] = WMMA_F16(s.v, b.v, o[nt]);
    }
  }

  // ---- store h_ret (f16) ----
  #pragma unroll
  for (int nt = 0; nt < 16; ++nt) {
    const int gc = nb + nt * 16 + (lane & 15);
    #pragma unroll
    for (int v = 0; v < 8; ++v) {
      const int gr = rowBase + q0 + (lane >> 4) * 8 + v;
      O[(size_t)gr * 512 + gc] = (_Float16)o[nt][v];
    }
  }
}

// ---------------------------------------------------------------------------
// PReLU + GroupNorm(G=16, 32 channels/group == wave32) + affine.
// One wave per row; lane l handles channel g*32+l of each group.
// ---------------------------------------------------------------------------
__global__ __launch_bounds__(256) void prelu_gn_kernel(
    const float* __restrict__ X, const float* __restrict__ pa,
    const float* __restrict__ gw, const float* __restrict__ gb,
    float* __restrict__ out)
{
  const int lane = threadIdx.x & 31;
  const int wid  = threadIdx.x >> 5;
  const int row  = blockIdx.x * 8 + wid;
  const float* xr = X + (size_t)row * 512;
  float* orow     = out + (size_t)row * 512;

  #pragma unroll
  for (int g = 0; g < 16; ++g) {
    const int c = g * 32 + lane;
    float t = xr[c];
    t = (t >= 0.0f) ? t : pa[c] * t;           // PReLU
    float s = t, q = t * t;
    #pragma unroll
    for (int off = 16; off > 0; off >>= 1) {   // 32-lane tree reduce
      s += __shfl_xor(s, off, 32);
      q += __shfl_xor(q, off, 32);
    }
    const float mean = s * (1.0f / 32.0f);
    const float var  = q * (1.0f / 32.0f) - mean * mean;
    const float xn   = (t - mean) * rsqrtf(var + 1e-5f);
    orow[c] = xn * gw[c] + gb[c];
  }
}

// ---------------------------------------------------------------------------
// Launch
// ---------------------------------------------------------------------------
extern "C" void kernel_launch(void* const* d_in, const int* in_sizes, int n_in,
                              void* d_out, int out_size, void* d_ws, size_t ws_size,
                              hipStream_t stream) {
  (void)in_sizes; (void)n_in; (void)out_size; (void)ws_size;

  const float* h       = (const float*)d_in[0];
  const float* D       = (const float*)d_in[1];
  const float* h_prime = (const float*)d_in[2];
  const float* Qw = (const float*)d_in[3];  const float* Qb = (const float*)d_in[4];
  const float* Kw = (const float*)d_in[5];  const float* Kb = (const float*)d_in[6];
  const float* Vw = (const float*)d_in[7];  const float* Vb = (const float*)d_in[8];
  const float* Rw = (const float*)d_in[9];  const float* Rb = (const float*)d_in[10];
  const float* Cw = (const float*)d_in[11]; const float* Cb = (const float*)d_in[12];
  const float* Pw = (const float*)d_in[13]; const float* Pb = (const float*)d_in[14];
  const float* pa = (const float*)d_in[15];
  const float* gw = (const float*)d_in[16];
  const float* gb = (const float*)d_in[17];

  // Workspace layout (bytes): Q,K,V,h_ret (8 MiB f16 each), combined (32 MiB f16), x (16 MiB f32)
  constexpr size_t SZ = (size_t)8192 * 512 * sizeof(_Float16);   // 8 MiB
  char* ws = (char*)d_ws;
  _Float16* Qf   = (_Float16*)(ws);
  _Float16* Kf   = (_Float16*)(ws + 1 * SZ);
  _Float16* Vf   = (_Float16*)(ws + 2 * SZ);
  _Float16* Hret = (_Float16*)(ws + 3 * SZ);
  _Float16* Comb = (_Float16*)(ws + 4 * SZ);                     // 8192 x 2048 f16
  float*    Xbuf = (float*)   (ws + 4 * SZ + (size_t)8192 * 2048 * sizeof(_Float16));

  const dim3 blk(256);

  // QKV projections: (8192x512) = h(8192x512) @ W(512x512) + b
  gemm_wmma_kernel<float, _Float16, false><<<dim3(4, 64), blk, 0, stream>>>(
      h, 512, Qw, 512, Qb, Qf, 512, 8192, 512, 512);
  gemm_wmma_kernel<float, _Float16, false><<<dim3(4, 64), blk, 0, stream>>>(
      h, 512, Kw, 512, Kb, Kf, 512, 8192, 512, 512);
  gemm_wmma_kernel<float, _Float16, false><<<dim3(4, 64), blk, 0, stream>>>(
      h, 512, Vw, 512, Vb, Vf, 512, 8192, 512, 512);

  // Fused decayed retention: h_ret = (D ∘ QK^T) V
  retention_kernel<<<dim3(128), blk, 0, stream>>>(Qf, Kf, Vf, D, Hret, 8192);

  // combined[:, 0:1024]   = gelu(h_ret @ Rw + Rb)
  gemm_wmma_kernel<_Float16, _Float16, true><<<dim3(8, 64), blk, 0, stream>>>(
      Hret, 512, Rw, 1024, Rb, Comb, 2048, 8192, 1024, 512);
  // combined[:, 1024:2048] = h_prime @ Cw + Cb
  gemm_wmma_kernel<float, _Float16, false><<<dim3(8, 64), blk, 0, stream>>>(
      h_prime, 512, Cw, 1024, Cb, Comb + 1024, 2048, 8192, 1024, 512);

  // x = combined @ Pw + Pb   (K = 2048)
  gemm_wmma_kernel<_Float16, float, false><<<dim3(4, 64), blk, 0, stream>>>(
      Comb, 2048, Pw, 512, Pb, Xbuf, 512, 8192, 512, 2048);

  // PReLU + GroupNorm + affine -> output (f32)
  prelu_gn_kernel<<<dim3(1024), blk, 0, stream>>>(Xbuf, pa, gw, gb, (float*)d_out);
}
